// GCN_2layer_76836964925933
// MI455X (gfx1250) — matrix operand order repaired and checked by reading, not compiled, not run on hardware
//
#include <hip/hip_runtime.h>

// ---------------------------------------------------------------------------
// 2-layer GeneralConv GNN for MI455X (gfx1250, wave32).
// All GEMMs (node linears + per-edge linear) run on v_wmma_f32_16x16x32_bf16
// with fully compile-time tiling; epilogues are guard-free (dims are exact
// multiples of the tile sizes: 50000%16==0, 800000%32==0), so EXEC stays full
// and no saveexec churn appears around stores/atomics.
// Scatter-add (segment_sum) via global_atomic_add_f32 into an L2-resident
// accumulator (51 MB << 192 MB L2). edge_attr is converted fp32->bf16 on the
// fly through LDS tile staging; weights are converted once per launch.
// Edge kernel uses 32-edge tiles so each B fragment feeds two 16-row WMMAs,
// halving weight re-reads from L2 (1.6 GB -> 0.8 GB for layer 1).
// ---------------------------------------------------------------------------

typedef __bf16 bf16_t;
typedef bf16_t v16bf __attribute__((ext_vector_type(16)));
typedef bf16_t v8bf  __attribute__((ext_vector_type(8)));
typedef bf16_t v4bf  __attribute__((ext_vector_type(4)));
typedef float  v8f   __attribute__((ext_vector_type(8)));

#define NN 50000
#define NE 800000
#define DI 64
#define DH 256
#define DO 64

// ---------------- weight fp32 -> bf16 conversion ----------------
__global__ void cvt_f32_bf16(const float* __restrict__ in, bf16_t* __restrict__ out, int n) {
  int i = blockIdx.x * blockDim.x + threadIdx.x;
  if (i < n) out[i] = (bf16_t)in[i];
}

// Build a 16x32 bf16 A-fragment (contiguous K order, matching B):
//   lane L: row L%16, K = kb..kb+15, kb = (L/16)*16
__device__ __forceinline__ v16bf load_afrag(const bf16_t* ldsRow, int kOff, int hk) {
  v8bf alo = *(const v8bf*)(ldsRow + kOff + hk * 16);
  v8bf ahi = *(const v8bf*)(ldsRow + kOff + hk * 16 + 8);
  return __builtin_shufflevector(alo, ahi,
      0, 1, 2, 3, 4, 5, 6, 7, 8, 9, 10, 11, 12, 13, 14, 15);
}

// ---------------- generic node GEMM: C[M,N] = act(A)[M,K] @ W[N,K]^T + b ----
// 128 threads = 4 waves; block tile = 16 rows x N cols. Per-wave n-tile count
// is compile-time (N=256 -> 4, N=64 -> 1); M must be a multiple of 16.
template<int K_TOTAL, int N, bool RELU_IN>
__global__ void gemm_bias(const float* __restrict__ A,
                          const bf16_t* __restrict__ W,
                          const float* __restrict__ bias,
                          float* __restrict__ C) {
  constexpr int NTILES = N / 16;
  constexpr int NT = NTILES / 4;          // n-tiles per wave
  static_assert(NTILES % 4 == 0, "N must be a multiple of 64");

  __shared__ __align__(16) bf16_t ldsA[16 * K_TOTAL];
  const int tid  = threadIdx.x;
  const int lane = tid & 31;
  const int wave = tid >> 5;
  const int m0   = blockIdx.x * 16;
  const int r    = lane & 15;
  const int hk   = lane >> 4;

  // stage full A strip [16 rows][K_TOTAL] fp32 -> bf16 (vectorized x4)
  constexpr int CHUNKS = 16 * K_TOTAL / 4;
#pragma unroll
  for (int idx = tid; idx < CHUNKS; idx += 128) {
    const int rr = idx / (K_TOTAL / 4);
    const int kk = (idx % (K_TOTAL / 4)) * 4;
    float4 v = *(const float4*)&A[(size_t)(m0 + rr) * K_TOTAL + kk];
    if (RELU_IN) {
      v.x = fmaxf(v.x, 0.f); v.y = fmaxf(v.y, 0.f);
      v.z = fmaxf(v.z, 0.f); v.w = fmaxf(v.w, 0.f);
    }
    v4bf o = { (bf16_t)v.x, (bf16_t)v.y, (bf16_t)v.z, (bf16_t)v.w };
    *(v4bf*)&ldsA[rr * K_TOTAL + kk] = o;
  }
  __syncthreads();

  v8f acc[NT] = {};
  const bf16_t* ldsRow = &ldsA[r * K_TOTAL];

#pragma unroll
  for (int k0 = 0; k0 < K_TOTAL; k0 += 32) {
    v16bf afrag = load_afrag(ldsRow, k0, hk);
#pragma unroll
    for (int t = 0; t < NT; ++t) {
      const int n0 = (wave + t * 4) * 16;
      // B fragment: lane L holds W row (n0 + L%16), K = k0 + (L/16)*16 .. +15
      const bf16_t* wp = W + (size_t)(n0 + r) * K_TOTAL + k0 + hk * 16;
      v16bf bfrag = *(const v16bf*)wp;
      acc[t] = __builtin_amdgcn_wmma_f32_16x16x32_bf16(
          false, afrag, false, bfrag, (short)0, acc[t], false, false);
    }
  }

#pragma unroll
  for (int t = 0; t < NT; ++t) {
    const int col = (wave + t * 4) * 16 + r;
    const float bv = bias[col];
#pragma unroll
    for (int v = 0; v < 8; ++v) {          // D elem v -> row m0 + v + 8*hk
      const int m = m0 + v + hk * 8;
      C[(size_t)m * N + col] = acc[t][v] + bv;
    }
  }
}

// ---------------- fused edge kernel -----------------------------------------
// Per block (128 thr / 4 waves): 32 edges (two 16-row M-subtiles sharing each
// B fragment).  t = edge_attr[e] @ We^T + be + node_msg[src[e]]
//               agg[dst[e]] += t   (f32 atomics at L2; accumulator resident)
template<int N_OUT>
__global__ void edge_conv(const float* __restrict__ edge_attr,  // [E,64]
                          const int* __restrict__ src,
                          const int* __restrict__ dst,
                          const bf16_t* __restrict__ W,         // [N_OUT,64]
                          const float* __restrict__ bias,       // [N_OUT]
                          const float* __restrict__ nmsg,       // [NN,N_OUT]
                          float* __restrict__ agg) {            // [NN,N_OUT]
  constexpr int NTILES = N_OUT / 16;
  constexpr int NT = NTILES / 4;
  static_assert(NTILES % 4 == 0, "N_OUT must be a multiple of 64");

  __shared__ __align__(16) bf16_t ldsA[32 * 64];
  __shared__ int lsrc[32], ldst[32];
  const int tid  = threadIdx.x;
  const int lane = tid & 31;
  const int wave = tid >> 5;
  const int e0   = blockIdx.x * 32;
  const int r    = lane & 15;
  const int hk   = lane >> 4;

  // speculative prefetch of the next block's contiguous 8KB edge tile
  // (global_prefetch_b8; dropped silently past the end of the buffer)
  __builtin_prefetch((const char*)&edge_attr[(size_t)(e0 + 32) * 64] + tid * 64, 0, 1);

  // stage 32 edges x 64 features, fp32 -> bf16 (vectorized x4)
#pragma unroll
  for (int idx = tid; idx < 32 * 64 / 4; idx += 128) {
    const int rr = idx >> 4;
    const int kk = (idx & 15) * 4;
    float4 v = *(const float4*)&edge_attr[(size_t)(e0 + rr) * 64 + kk];
    v4bf o = { (bf16_t)v.x, (bf16_t)v.y, (bf16_t)v.z, (bf16_t)v.w };
    *(v4bf*)&ldsA[rr * 64 + kk] = o;
  }
  if (tid < 32) {
    lsrc[tid] = src[e0 + tid];
    ldst[tid] = dst[e0 + tid];
  }
  __syncthreads();

  v8f acc0[NT] = {};
  v8f acc1[NT] = {};
  const bf16_t* ldsRow0 = &ldsA[r * 64];
  const bf16_t* ldsRow1 = &ldsA[(16 + r) * 64];

#pragma unroll
  for (int k0 = 0; k0 < 64; k0 += 32) {
    v16bf af0 = load_afrag(ldsRow0, k0, hk);
    v16bf af1 = load_afrag(ldsRow1, k0, hk);
#pragma unroll
    for (int t = 0; t < NT; ++t) {
      const int n0 = (wave + t * 4) * 16;
      const bf16_t* wp = W + (size_t)(n0 + r) * 64 + k0 + hk * 16;
      v16bf bfrag = *(const v16bf*)wp;   // shared by both M-subtiles
      acc0[t] = __builtin_amdgcn_wmma_f32_16x16x32_bf16(
          false, af0, false, bfrag, (short)0, acc0[t], false, false);
      acc1[t] = __builtin_amdgcn_wmma_f32_16x16x32_bf16(
          false, af1, false, bfrag, (short)0, acc1[t], false, false);
    }
  }

#pragma unroll
  for (int t = 0; t < NT; ++t) {
    const int col = (wave + t * 4) * 16 + r;
    const float bv = bias[col];
#pragma unroll
    for (int v = 0; v < 8; ++v) {
      const int m = v + hk * 8;                 // row within subtile
      {
        const int s = lsrc[m], d = ldst[m];
        float val = acc0[t][v] + bv + nmsg[(size_t)s * N_OUT + col];
        atomicAdd(&agg[(size_t)d * N_OUT + col], val);
      }
      {
        const int s = lsrc[16 + m], d = ldst[16 + m];
        float val = acc1[t][v] + bv + nmsg[(size_t)s * N_OUT + col];
        atomicAdd(&agg[(size_t)d * N_OUT + col], val);
      }
    }
  }
}

__global__ void relu_inplace(float* __restrict__ x, int n) {
  int i = blockIdx.x * blockDim.x + threadIdx.x;
  if (i < n) x[i] = fmaxf(x[i], 0.0f);
}

// ---------------------------------------------------------------------------
extern "C" void kernel_launch(void* const* d_in, const int* in_sizes, int n_in,
                              void* d_out, int out_size, void* d_ws, size_t ws_size,
                              hipStream_t stream) {
  const float* x          = (const float*)d_in[0];
  const int*   edge_index = (const int*)d_in[1];   // [2, E]
  const float* edge_attr  = (const float*)d_in[2];
  const float* W1m = (const float*)d_in[3];  const float* b1m = (const float*)d_in[4];
  const float* W1e = (const float*)d_in[5];  const float* b1e = (const float*)d_in[6];
  const float* W1s = (const float*)d_in[7];  const float* b1s = (const float*)d_in[8];
  const float* W2m = (const float*)d_in[9];  const float* b2m = (const float*)d_in[10];
  const float* W2e = (const float*)d_in[11]; const float* b2e = (const float*)d_in[12];
  const float* W2s = (const float*)d_in[13]; const float* b2s = (const float*)d_in[14];
  const int* src = edge_index;
  const int* dst = edge_index + NE;
  float* out = (float*)d_out;

  // workspace layout
  char* ws = (char*)d_ws;
  float* xmsg = (float*)ws;                                  // [NN, DH] (reused as hmsg [NN, DO])
  float* agg1 = (float*)(ws + (size_t)NN * DH * 4);          // [NN, DH]
  bf16_t* wb  = (bf16_t*)(ws + (size_t)2 * NN * DH * 4);
  bf16_t* W1m_b = wb;                      // 256x64
  bf16_t* W1e_b = W1m_b + DH * DI;
  bf16_t* W1s_b = W1e_b + DH * DI;
  bf16_t* W2m_b = W1s_b + DH * DI;         // 64x256
  bf16_t* W2e_b = W2m_b + DO * DH;         // 64x64
  bf16_t* W2s_b = W2e_b + DO * DI;         // 64x256

  auto cvt = [&](const float* in, bf16_t* o, int n) {
    cvt_f32_bf16<<<(n + 255) / 256, 256, 0, stream>>>(in, o, n);
  };
  cvt(W1m, W1m_b, DH * DI);
  cvt(W1e, W1e_b, DH * DI);
  cvt(W1s, W1s_b, DH * DI);
  cvt(W2m, W2m_b, DO * DH);
  cvt(W2e, W2e_b, DO * DI);
  cvt(W2s, W2s_b, DO * DH);

  const int mblocks = NN / 16;   // 3125 (exact)
  const int eblocks = NE / 32;   // 25000 (exact)

  // ---- layer 1 ----
  gemm_bias<64, DH, false><<<mblocks, 128, 0, stream>>>(x, W1s_b, b1s, agg1);
  gemm_bias<64, DH, false><<<mblocks, 128, 0, stream>>>(x, W1m_b, b1m, xmsg);
  edge_conv<DH><<<eblocks, 128, 0, stream>>>(edge_attr, src, dst, W1e_b, b1e,
                                             xmsg, agg1);

  // ---- layer 2 (ReLU on h folded into A-load) ----
  gemm_bias<256, DO, true><<<mblocks, 128, 0, stream>>>(agg1, W2s_b, b2s, out);
  gemm_bias<256, DO, true><<<mblocks, 128, 0, stream>>>(agg1, W2m_b, b2m, xmsg);
  edge_conv<DO><<<eblocks, 128, 0, stream>>>(edge_attr, src, dst, W2e_b, b2e,
                                             xmsg, out);

  relu_inplace<<<((NN * DO) + 255) / 256, 256, 0, stream>>>(out, NN * DO);
}
